// Encoder_28887950033670
// MI455X (gfx1250) — matrix operand order repaired and checked by reading, not compiled, not run on hardware
//
#include <hip/hip_runtime.h>
#include <hip/hip_bf16.h>
#include <stdint.h>

// ---------------- types for WMMA ----------------
typedef __attribute__((ext_vector_type(16))) __bf16 v16bf;
typedef __attribute__((ext_vector_type(8)))  float  v8f;

static __device__ __forceinline__ uint16_t f2bf(float f) {
    uint32_t x = __float_as_uint(f);
    uint32_t r = x + 0x7fffu + ((x >> 16) & 1u);   // round-to-nearest-even
    return (uint16_t)(r >> 16);
}

static __device__ __forceinline__ float elu(float v) {
    return v > 0.0f ? v : (__expf(v) - 1.0f);
}

// ---------------- degree / norm ----------------
__global__ void deg_init_kernel(float* deg, int n) {
    int i = blockIdx.x * blockDim.x + threadIdx.x;
    if (i < n) deg[i] = 1.0f;                      // self loop
}
__global__ void deg_count_kernel(const int* __restrict__ dst, float* deg, int E) {
    int e = blockIdx.x * blockDim.x + threadIdx.x;
    if (e < E) atomicAdd(&deg[dst[e]], 1.0f);
}
__global__ void dinv_kernel(float* deg, int n) {
    int i = blockIdx.x * blockDim.x + threadIdx.x;
    if (i < n) deg[i] = __frsqrt_rn(deg[i]);       // deg >= 1 always
}

// ---------------- conversions / packing ----------------
__global__ void conv_bf16_kernel(const float* __restrict__ x, uint16_t* __restrict__ a, long n) {
    long i = (long)blockIdx.x * blockDim.x + threadIdx.x;
    if (i < n) a[i] = f2bf(x[i]);
}
// Wp[n][k/2] dword = { bf16(W[k][n]) , bf16(W[k+1][n])<<16 }  (K = rows of W, Dout = cols)
__global__ void pack_w_kernel(const float* __restrict__ W, uint32_t* __restrict__ Wp,
                              int K, int Dout) {
    int idx = blockIdx.x * blockDim.x + threadIdx.x;
    int K2 = K >> 1;
    if (idx >= Dout * K2) return;
    int nC = idx / K2, k2 = idx - nC * K2;
    uint32_t lo = f2bf(W[(size_t)(2 * k2) * Dout + nC]);
    uint32_t hi = f2bf(W[(size_t)(2 * k2 + 1) * Dout + nC]);
    Wp[(size_t)nC * K2 + k2] = lo | (hi << 16);
}

// ---------------- WMMA GEMM: out[M x Dout] = Abf[M x 512] * W ----------------
// one wave -> 16(M) x 64(N) tile; K fixed at 512; register double-buffered
__global__ void wmma_gemm_bf16_kernel(const uint32_t* __restrict__ A,   // bf16 pairs, row-major [M][K/2]
                                      const uint32_t* __restrict__ Wp,  // packed [Dout][K/2]
                                      float* __restrict__ out,
                                      int Mtiles, int Dout) {
    constexpr int K     = 512;
    constexpr int K2    = K / 2;
    constexpr int KSTEP = K / 32;                  // 16
    const int ntile64 = Dout >> 6;
    const int wave = (blockIdx.x * blockDim.x + threadIdx.x) >> 5;
    if (wave >= Mtiles * ntile64) return;          // whole-wave uniform exit
    const int lane = threadIdx.x & 31;
    const int hf   = lane >> 4;                    // half-wave select
    const int lm   = lane & 15;
    const int tileM = wave / ntile64;
    const int nbase = (wave - tileM * ntile64) << 6;

    const uint4* Arow = (const uint4*)(A + (size_t)(tileM * 16 + lm) * K2);
    const uint4* Brow[4];
#pragma unroll
    for (int s = 0; s < 4; ++s)
        Brow[s] = (const uint4*)(Wp + (size_t)(nbase + s * 16 + lm) * K2);

    union V16 { uint4 q[2]; v16bf v; };
    v8f acc[4] = {};
    V16 a[2];
    V16 b[2][4];

    // prologue: stage k-step 0
    // A 16x32 bf16: VGPR 0-3 = K {0,2,4,6}+8*hf ; VGPR 4-7 = K {16,18,20,22}+8*hf
    a[0].q[0] = Arow[hf];
    a[0].q[1] = Arow[2 + hf];
#pragma unroll
    for (int s = 0; s < 4; ++s) {
        // B 32x16 bf16: lane = col N, VGPR 0-3 = K {0,2,4,6}+8*hf ; VGPR 4-7 = K {16,18,20,22}+8*hf
        b[0][s].q[0] = Brow[s][hf];
        b[0][s].q[1] = Brow[s][2 + hf];
    }

#pragma unroll
    for (int kt = 0; kt < KSTEP; ++kt) {
        const int cur = kt & 1, nxt = cur ^ 1;
        if (kt + 1 < KSTEP) {
            // issue next k-step's 10 b128 loads; they drain behind the 4 WMMAs below
            a[nxt].q[0] = Arow[4 * (kt + 1) + hf];
            a[nxt].q[1] = Arow[4 * (kt + 1) + 2 + hf];
#pragma unroll
            for (int s = 0; s < 4; ++s) {
                b[nxt][s].q[0] = Brow[s][4 * (kt + 1) + hf];
                b[nxt][s].q[1] = Brow[s][4 * (kt + 1) + 2 + hf];
            }
        }
        if (kt + 2 < KSTEP)
            __builtin_prefetch((const void*)&Arow[4 * (kt + 2) + hf], 0, 1);
#pragma unroll
        for (int s = 0; s < 4; ++s)
            acc[s] = __builtin_amdgcn_wmma_f32_16x16x32_bf16(
                false, a[cur].v, false, b[cur][s].v, (short)0, acc[s], false, false);
    }

    // C/D: VGPR r -> row M = r + 8*hf, lane -> col
#pragma unroll
    for (int s = 0; s < 4; ++s) {
        float* orow = out + (size_t)(tileM * 16 + 8 * hf) * Dout + nbase + s * 16 + lm;
#pragma unroll
        for (int r = 0; r < 8; ++r)
            orow[(size_t)r * Dout] = acc[s][r];
    }
}

// ---------------- aggregation ----------------
// AGG[i,:] = H[i,:] * dinv[i]^2   (self loop contribution)
__global__ void self_init_kernel(const float* __restrict__ H, const float* __restrict__ dinv,
                                 float* __restrict__ AGG, int n, int D) {
    long idx = (long)blockIdx.x * blockDim.x + threadIdx.x;
    int D4 = D >> 2;
    if (idx >= (long)n * D4) return;
    int i = (int)(idx / D4);
    int c = (int)(idx - (long)i * D4) << 2;
    float w = dinv[i] * dinv[i];
    const float4 hv = *(const float4*)(H + (size_t)i * D + c);
    float4 o; o.x = hv.x * w; o.y = hv.y * w; o.z = hv.z * w; o.w = hv.w * w;
    *(float4*)(AGG + (size_t)i * D + c) = o;
}

// AGG[dst,:] += H[src,:] * dinv[src]*dinv[dst]
__global__ void scatter_kernel(const int* __restrict__ src, const int* __restrict__ dst,
                               const float* __restrict__ dinv, const float* __restrict__ H,
                               float* __restrict__ AGG, int E, int D) {
    long idx = (long)blockIdx.x * blockDim.x + threadIdx.x;
    int D4 = D >> 2;
    if (idx >= (long)E * D4) return;
    int e = (int)(idx / D4);
    int c = (int)(idx - (long)e * D4) << 2;
    int s = src[e], d = dst[e];
    float nrm = dinv[s] * dinv[d];
    const float4 hv = *(const float4*)(H + (size_t)s * D + c);
    float* o = AGG + (size_t)d * D + c;
    atomicAdd(o + 0, hv.x * nrm);
    atomicAdd(o + 1, hv.y * nrm);
    atomicAdd(o + 2, hv.z * nrm);
    atomicAdd(o + 3, hv.w * nrm);
}

// layer-1 epilogue: Abf[i,c] = bf16( elu(AGG[i,c] + b[c]) )
__global__ void post1_kernel(const float* __restrict__ AGG, const float* __restrict__ b,
                             uint16_t* __restrict__ Abf, int n, int D) {
    long idx = (long)blockIdx.x * blockDim.x + threadIdx.x;
    if (idx >= (long)n * D) return;
    int c = (int)(idx % D);
    Abf[idx] = f2bf(elu(AGG[idx] + b[c]));
}

// layer-2 epilogue (in place on d_out): out = elu(out + b[c])
__global__ void post2_kernel(float* __restrict__ out, const float* __restrict__ b,
                             int n, int D) {
    long idx = (long)blockIdx.x * blockDim.x + threadIdx.x;
    if (idx >= (long)n * D) return;
    int c = (int)(idx % D);
    out[idx] = elu(out[idx] + b[c]);
}

// ---------------- launcher ----------------
extern "C" void kernel_launch(void* const* d_in, const int* in_sizes, int n_in,
                              void* d_out, int out_size, void* d_ws, size_t ws_size,
                              hipStream_t stream) {
    constexpr int D_IN = 512, D_H = 512, D_OUT = 256;
    const float* x  = (const float*)d_in[0];
    const int*   ei = (const int*)d_in[1];
    const float* W1 = (const float*)d_in[2];
    const float* b1 = (const float*)d_in[3];
    const float* W2 = (const float*)d_in[4];
    const float* b2 = (const float*)d_in[5];
    float* outp = (float*)d_out;

    const int n = in_sizes[0] / D_IN;      // 50000
    const int E = in_sizes[1] / 2;         // 500000
    const int* src = ei;
    const int* dst = ei + E;

    // workspace layout
    char* ws = (char*)d_ws;
    size_t off = 0;
    auto carve = [&](size_t bytes) -> void* {
        void* p = ws + off;
        off += (bytes + 255) & ~(size_t)255;
        return p;
    };
    float*    dinv = (float*)   carve((size_t)n * 4);
    uint16_t* ABF  = (uint16_t*)carve((size_t)n * D_IN * 2);        // A for both layers
    uint32_t* Wp1  = (uint32_t*)carve((size_t)D_H * (D_IN / 2) * 4);
    uint32_t* Wp2  = (uint32_t*)carve((size_t)D_OUT * (D_H / 2) * 4);
    float*    H    = (float*)   carve((size_t)n * D_H * 4);          // H1, reused as H2
    float*    AGG  = (float*)   carve((size_t)n * D_H * 4);

    const int TB = 256;
    auto blocks = [](long work, int tb) { return (int)((work + tb - 1) / tb); };

    // degrees -> dinv (in place)
    deg_init_kernel<<<blocks(n, TB), TB, 0, stream>>>(dinv, n);
    deg_count_kernel<<<blocks(E, TB), TB, 0, stream>>>(dst, dinv, E);
    dinv_kernel<<<blocks(n, TB), TB, 0, stream>>>(dinv, n);

    // bf16 conversion / weight packing
    conv_bf16_kernel<<<blocks((long)n * D_IN, TB), TB, 0, stream>>>(x, ABF, (long)n * D_IN);
    pack_w_kernel<<<blocks((long)D_H * (D_IN / 2), TB), TB, 0, stream>>>(W1, Wp1, D_IN, D_H);
    pack_w_kernel<<<blocks((long)D_OUT * (D_H / 2), TB), TB, 0, stream>>>(W2, Wp2, D_H, D_OUT);

    // ---- layer 1 ----
    const int Mtiles = n / 16;  // 3125
    {
        long waves = (long)Mtiles * (D_H / 64);
        wmma_gemm_bf16_kernel<<<blocks(waves * 32, TB), TB, 0, stream>>>(
            (const uint32_t*)ABF, Wp1, H, Mtiles, D_H);
    }
    self_init_kernel<<<blocks((long)n * (D_H / 4), TB), TB, 0, stream>>>(H, dinv, AGG, n, D_H);
    scatter_kernel<<<blocks((long)E * (D_H / 4), TB), TB, 0, stream>>>(src, dst, dinv, H, AGG, E, D_H);
    post1_kernel<<<blocks((long)n * D_H, TB), TB, 0, stream>>>(AGG, b1, ABF, n, D_H);

    // ---- layer 2 ----
    {
        long waves = (long)Mtiles * (D_OUT / 64);
        wmma_gemm_bf16_kernel<<<blocks(waves * 32, TB), TB, 0, stream>>>(
            (const uint32_t*)ABF, Wp2, H, Mtiles, D_OUT);
    }
    self_init_kernel<<<blocks((long)n * (D_OUT / 4), TB), TB, 0, stream>>>(H, dinv, outp, n, D_OUT);
    scatter_kernel<<<blocks((long)E * (D_OUT / 4), TB), TB, 0, stream>>>(src, dst, dinv, H, outp, E, D_OUT);
    post2_kernel<<<blocks((long)n * D_OUT, TB), TB, 0, stream>>>(outp, b2, n, D_OUT);
}
